// MetaGRUCell_38714835206788
// MI455X (gfx1250) — matrix operand order repaired and checked by reading, not compiled, not run on hardware
//
#include <hip/hip_runtime.h>
#include <hip/hip_bf16.h>

// MetaGRU for MI455X (gfx1250), wave32 + WMMA f32_16x16x32_f16.
// One workgroup per node: per-node weights generated once into LDS (f16),
// hidden state kept in LDS across the T-loop, all matmuls via v_wmma.
// Rows are partitioned per-wave -> per-timestep sync is just s_wait_dscnt.

#define F_IN 32
#define HID  64
#define FEAT 32
#define B_   64
#define T_   12
#define N_   1000

typedef __attribute__((ext_vector_type(16))) _Float16 v16h;
typedef __attribute__((ext_vector_type(8)))  _Float16 v8h;
typedef __attribute__((ext_vector_type(8)))  float    v8f;

union AFrag { v16h v; v8h h[2]; };

// A-fragment (16x32 f16, row-major source, row stride lda):
// lane L<16 : row M=m0+L,   K = k0+{0..7} then k0+{16..23}
// lane L>=16: row M=m0+L-16,K = k0+{8..15} then k0+{24..31}
__device__ __forceinline__ v16h load_frag_a(const _Float16* base, int lda,
                                            int m0, int k0, int laneLo, int laneHi) {
  const _Float16* p = base + (m0 + laneLo) * lda + k0 + laneHi * 8;
  AFrag f;
  f.h[0] = *(const v8h*)(p);
  f.h[1] = *(const v8h*)(p + 16);
  return f.v;
}

// B-fragment (32x16 f16). Weights stored transposed: WT[out][k], row stride ldb.
// lane L<16 : col N=n0+L, K = k0+{0..7} then k0+{16..23}; lane L>=16: K = k0+{8..15} then k0+{24..31}.
__device__ __forceinline__ v16h load_frag_b(const _Float16* wt, int ldb,
                                            int n0, int k0, int laneLo, int laneHi) {
  const _Float16* p = wt + (n0 + laneLo) * ldb + k0 + laneHi * 8;
  AFrag f;
  f.h[0] = *(const v8h*)(p);
  f.h[1] = *(const v8h*)(p + 16);
  return f.v;
}

__device__ __forceinline__ v8f wmma_f16(v16h a, v16h b, v8f c) {
  return __builtin_amdgcn_wmma_f32_16x16x32_f16(false, a, false, b,
                                                (short)0, c, false, false);
}

// Fast transcendentals: v_exp_f32 + raw v_rcp_f32 (no IEEE divide expansion).
__device__ __forceinline__ float sigmoidf_(float x) {
  return __builtin_amdgcn_rcpf(1.0f + __expf(-x));
}
__device__ __forceinline__ float tanhf_(float x) {
  return 1.0f - 2.0f * __builtin_amdgcn_rcpf(__expf(2.0f * x) + 1.0f);
}

// Per-wave LDS ordering: all row data touched by a wave is wave-private, so a
// dscnt drain + scheduling fence replaces block-wide s_barrier.
__device__ __forceinline__ void lds_fence_wave() {
  __builtin_amdgcn_wave_barrier();
  asm volatile("s_wait_dscnt 0x0" ::: "memory");
  __builtin_amdgcn_wave_barrier();
}

struct SetPtrs {  // alphabetical key order of the param dicts
  const float *b1, *b2, *b3, *bb1, *bb2, *bw1, *bw2, *w1, *w2, *w3;
};
struct AllParams { SetPtrs s[4]; };  // 0:z 1:r 2:i2h 3:h2h

// LDS layout (dynamic): f16 region then f32 region, all 16B aligned.
#define SM_H16 (64*96 + 64*96 + 64*32 + 64*64 + 64*96 + 64*64)  // 28672 halves
#define SM_F32 (64*64 + 4*64 + 4*32 + 8 + 8 + 32)               // 4528 floats
#define SMEM_BYTES (SM_H16 * 2 + SM_F32 * 4)                    // 75456 B

__global__ __launch_bounds__(128)
void meta_gru_kernel(const float* __restrict__ feature,
                     const float* __restrict__ data,
                     AllParams P,
                     float* __restrict__ out,
                     float* __restrict__ state)
{
  extern __shared__ char smem_raw[];
  _Float16* WzT = (_Float16*)smem_raw;      // [64][96]
  _Float16* WrT = WzT + 64 * 96;            // [64][96]
  _Float16* WiT = WrT + 64 * 96;            // [64][32]
  _Float16* WhT = WiT + 64 * 32;            // [64][64]
  _Float16* dsb = WhT + 64 * 64;            // [64][96]  ds = [x_t | h] f16
  _Float16* rhb = dsb + 64 * 96;            // [64][64]  r*h f16
  float* hbuf = (float*)(rhb + 64 * 64);    // [64][64]  hidden state f32
  float* bias = hbuf + 64 * 64;             // [4][64]   bz,br,bi,bh
  float* h1b  = bias + 4 * 64;              // [4][32]
  float* h2b  = h1b + 4 * 32;               // [4][2]
  float* gbuf = h2b + 8;                    // [4][2]
  float* featl = gbuf + 8;                  // [32]

  const int tid = threadIdx.x;
  const int node = blockIdx.x;
  const int lane = tid & 31;
  const int wave = tid >> 5;
  const int laneLo = lane & 15;
  const int laneHi = lane >> 4;
  const int mBase = wave * 16;

  // ---------------- weight generation (once per node) ----------------
  if (tid < FEAT) featl[tid] = feature[(size_t)node * FEAT + tid];
  for (int i = tid; i < 64 * 64; i += 128) hbuf[i] = 0.0f;
  __syncthreads();

  {  // h1 = sigmoid(feat @ w1 + b1): 4 sets x 32 units, one per thread
    int s = tid >> 5, j = tid & 31;
    const SetPtrs& p = P.s[s];
    float acc = p.b1[j];
    for (int i = 0; i < FEAT; ++i) acc += featl[i] * p.w1[i * 32 + j];
    h1b[s * 32 + j] = sigmoidf_(acc);
  }
  __syncthreads();
  if (tid < 8) {  // h2 = sigmoid(h1 @ w2 + b2)
    int s = tid >> 1, j2 = tid & 1;
    const SetPtrs& p = P.s[s];
    float acc = p.b2[j2];
    for (int j = 0; j < 32; ++j) acc += h1b[s * 32 + j] * p.w2[j * 2 + j2];
    h2b[s * 2 + j2] = sigmoidf_(acc);
  } else if (tid < 16) {  // g = sigmoid(feat @ bw1 + bb1)
    int s = (tid - 8) >> 1, j2 = tid & 1;
    const SetPtrs& p = P.s[s];
    float acc = p.bb1[j2];
    for (int i = 0; i < FEAT; ++i) acc += featl[i] * p.bw1[i * 2 + j2];
    gbuf[s * 2 + j2] = sigmoidf_(acc);
  }
  __syncthreads();

  // biases: b[s][o] = g0*bw2[0][o] + g1*bw2[1][o] + bb2[o]
  for (int idx = tid; idx < 4 * 64; idx += 128) {
    int s = idx >> 6, o = idx & 63;
    const SetPtrs& p = P.s[s];
    bias[idx] = gbuf[s * 2] * p.bw2[o] + gbuf[s * 2 + 1] * p.bw2[64 + o] + p.bb2[o];
  }
  // weights: W[k][o] = h2_0*w3[0][k*64+o] + h2_1*w3[1][k*64+o] + b3 -> WT[o][k] f16
  {
    const int Ks[4] = {96, 96, 32, 64};
    _Float16* const WT[4] = {WzT, WrT, WiT, WhT};
#pragma unroll
    for (int s = 0; s < 4; ++s) {
      const SetPtrs& p = P.s[s];
      const int K = Ks[s];
      const float g0 = h2b[s * 2], g1 = h2b[s * 2 + 1];
      for (int idx = tid; idx < K * 64; idx += 128) {
        int k = idx >> 6, o = idx & 63;
        float w = g0 * p.w3[idx] + g1 * p.w3[K * 64 + idx] + p.b3[idx];
        WT[s][o * K + k] = (_Float16)w;
      }
    }
  }
  __syncthreads();   // weights/biases now visible to all waves; h zeroed

  // hoist loop-invariant biases into registers (column o = nt*16+laneLo)
  float bzv[4], brv[4], biv[4], bhv[4];
#pragma unroll
  for (int nt = 0; nt < 4; ++nt) {
    const int o = nt * 16 + laneLo;
    bzv[nt] = bias[o];
    brv[nt] = bias[64 + o];
    biv[nt] = bias[128 + o];
    bhv[nt] = bias[192 + o];
  }

  // ---------------- recurrent scan over T ----------------
  for (int t = 0; t < T_; ++t) {
    // stage ds = [x_t | h] as f16 (each wave stages its own 16 rows; lane=col)
#pragma unroll 4
    for (int r = 0; r < 16; ++r) {
      const int m = mBase + r;
      float xv = data[((((size_t)m * T_) + t) * N_ + node) * F_IN + lane];
      dsb[m * 96 + lane] = (_Float16)xv;
      dsb[m * 96 + 32 + lane] = (_Float16)hbuf[m * 64 + lane];
      dsb[m * 96 + 64 + lane] = (_Float16)hbuf[m * 64 + 32 + lane];
    }
    lds_fence_wave();

    v8f zacc[4], cacc[4];
    const v16h a0 = load_frag_a(dsb, 96, mBase, 0,  laneLo, laneHi);
    const v16h a1 = load_frag_a(dsb, 96, mBase, 32, laneLo, laneHi);
    const v16h a2 = load_frag_a(dsb, 96, mBase, 64, laneLo, laneHi);

#pragma unroll
    for (int nt = 0; nt < 4; ++nt) {
      v8f z = {}, r = {}, c = {};
      z = wmma_f16(a0, load_frag_b(WzT, 96, nt * 16, 0,  laneLo, laneHi), z);
      z = wmma_f16(a1, load_frag_b(WzT, 96, nt * 16, 32, laneLo, laneHi), z);
      z = wmma_f16(a2, load_frag_b(WzT, 96, nt * 16, 64, laneLo, laneHi), z);
      r = wmma_f16(a0, load_frag_b(WrT, 96, nt * 16, 0,  laneLo, laneHi), r);
      r = wmma_f16(a1, load_frag_b(WrT, 96, nt * 16, 32, laneLo, laneHi), r);
      r = wmma_f16(a2, load_frag_b(WrT, 96, nt * 16, 64, laneLo, laneHi), r);
      c = wmma_f16(a0, load_frag_b(WiT, 32, nt * 16, 0,  laneLo, laneHi), c);

      const int o = nt * 16 + laneLo;
#pragma unroll
      for (int v = 0; v < 8; ++v) {
        const int m = mBase + v + laneHi * 8;
        float zz = sigmoidf_(z[v] + bzv[nt]);
        float rr = sigmoidf_(r[v] + brv[nt]);
        float cc = tanhf_(c[v] + biv[nt]);
        float hv = hbuf[m * 64 + o];
        rhb[m * 64 + o] = (_Float16)(rr * hv);
        z[v] = zz;
        c[v] = cc;
      }
      zacc[nt] = z;
      cacc[nt] = c;
    }
    lds_fence_wave();

#pragma unroll
    for (int nt = 0; nt < 4; ++nt) {
      v8f hh = {};
      hh = wmma_f16(load_frag_a(rhb, 64, mBase, 0,  laneLo, laneHi),
                    load_frag_b(WhT, 64, nt * 16, 0,  laneLo, laneHi), hh);
      hh = wmma_f16(load_frag_a(rhb, 64, mBase, 32, laneLo, laneHi),
                    load_frag_b(WhT, 64, nt * 16, 32, laneLo, laneHi), hh);

      const int o = nt * 16 + laneLo;
#pragma unroll
      for (int v = 0; v < 8; ++v) {
        const int m = mBase + v + laneHi * 8;
        float hv = hbuf[m * 64 + o];
        float zz = zacc[nt][v], cc = cacc[nt][v];
        float hn = zz * hv + (1.0f - zz) * cc + (hh[v] + bhv[nt]);
        hbuf[m * 64 + o] = hn;
      }
    }
    lds_fence_wave();

    // coalesced b128 output stores of this wave's 16 rows from hbuf
#pragma unroll 2
    for (int idx = lane; idx < 16 * 16; idx += 32) {  // 16 rows x 16 float4
      const int r = idx >> 4, c4 = idx & 15;
      const int m = mBase + r;
      float4 v = *(const float4*)&hbuf[m * 64 + c4 * 4];
      *(float4*)&out[((((size_t)m * T_) + t) * N_ + node) * HID + c4 * 4] = v;
    }
  }

  // final state = last h (wave-private rows, b128 stores)
  for (int idx = lane; idx < 16 * 16; idx += 32) {
    const int r = idx >> 4, c4 = idx & 15;
    const int m = mBase + r;
    float4 v = *(const float4*)&hbuf[m * 64 + c4 * 4];
    *(float4*)&state[((size_t)m * N_ + node) * HID + c4 * 4] = v;
  }
}

extern "C" void kernel_launch(void* const* d_in, const int* in_sizes, int n_in,
                              void* d_out, int out_size, void* d_ws, size_t ws_size,
                              hipStream_t stream) {
  (void)in_sizes; (void)n_in; (void)out_size; (void)d_ws; (void)ws_size;
  const float* feature = (const float*)d_in[0];
  const float* data    = (const float*)d_in[1];

  AllParams P;
  // param dicts flattened in alphabetical key order:
  // b1,b2,b3,bb1,bb2,bw1,bw2,w1,w2,w3 ; sets in setup order: z, r, i2h, h2h
  for (int s = 0; s < 4; ++s) {
    const float* const* q = (const float* const*)(d_in + 2 + s * 10);
    P.s[s].b1  = q[0]; P.s[s].b2  = q[1]; P.s[s].b3  = q[2];
    P.s[s].bb1 = q[3]; P.s[s].bb2 = q[4]; P.s[s].bw1 = q[5];
    P.s[s].bw2 = q[6]; P.s[s].w1  = q[7]; P.s[s].w2  = q[8]; P.s[s].w3 = q[9];
  }

  float* out   = (float*)d_out;                          // [B,T,N,HID]
  float* state = out + (size_t)B_ * T_ * N_ * HID;       // [B,N,HID]

  meta_gru_kernel<<<N_, 128, SMEM_BYTES, stream>>>(feature, data, P, out, state);
}
